// ChannelizedLinearCompression_1039382086303
// MI455X (gfx1250) — matrix operand order repaired and checked by reading, not compiled, not run on hardware
//
#include <hip/hip_runtime.h>

typedef __attribute__((ext_vector_type(16))) __bf16 v16bf;
typedef __attribute__((ext_vector_type(8)))  float  v8f;
typedef __attribute__((ext_vector_type(4)))  float  f32x4;
typedef __attribute__((ext_vector_type(2)))  __bf16 bf16x2;
typedef __attribute__((ext_vector_type(4)))  __bf16 bf16x4;
typedef __attribute__((ext_vector_type(4)))  int    v4i;

#define B_SZ   2048
#define C_SZ   12
#define N_SZ   8192
#define H_SZ   286
#define HPAD   288
#define MID    10
#define FH     30
#define LD     16

#define BM     64          // M rows per workgroup
#define KC     32          // K per WMMA chunk (bf16)
#define LDA_S  40          // padded LDS row stride (elems) for A
#define LDB_S  40          // padded LDS col stride (elems) for B
#define NK     (N_SZ / KC)

union FragU { uint4 u4[2]; v16bf v; };

static __device__ __forceinline__ void split_bf16(float v, __bf16 &hi, __bf16 &lo) {
  hi = (__bf16)v;
  lo = (__bf16)(v - (float)hi);
}

// ---- CDNA5 async global->LDS copy (ASYNCcnt path), guarded by __has_builtin ----
#if __has_builtin(__builtin_amdgcn_global_load_async_to_lds_b128)
#define HAVE_ASYNC 1
#endif

typedef __attribute__((address_space(1))) v4i gv4i;
typedef __attribute__((address_space(3))) v4i lv4i;

static __device__ __forceinline__ void async_cp16(__bf16* dst_lds, const __bf16* src_g) {
#ifdef HAVE_ASYNC
  __builtin_amdgcn_global_load_async_to_lds_b128(
      (gv4i*)(src_g), (lv4i*)(dst_lds), 0, 0);
#else
  *(v4i*)dst_lds = *(const v4i*)src_g;
#endif
}

static __device__ __forceinline__ void wait_async() {
#ifdef HAVE_ASYNC
#if __has_builtin(__builtin_amdgcn_s_wait_asynccnt)
  __builtin_amdgcn_s_wait_asynccnt(0);
#else
  asm volatile("s_wait_asynccnt 0x0" ::: "memory");
#endif
#endif
}

// ---------------------------------------------------------------------------
// Kernel 0: W1 [c][n][h] f32  ->  W1T hi/lo bf16 planes [c][h_pad=288][n]
// ---------------------------------------------------------------------------
__global__ void __launch_bounds__(256) w1_prep(const float* __restrict__ W1,
                                               __bf16* __restrict__ hiP,
                                               __bf16* __restrict__ loP) {
  __shared__ float tile[32][33];
  const int n0 = blockIdx.x * 32;
  const int hb0 = blockIdx.y * 32;
  const int c  = blockIdx.z;
  const int tid = threadIdx.x;
  #pragma unroll
  for (int i = 0; i < 4; ++i) {
    int id = tid + i * 256;
    int in = id >> 5, j = id & 31;
    int h = hb0 + j;
    float v = 0.f;
    if (h < H_SZ) v = W1[((long)(c * N_SZ + n0 + in)) * H_SZ + h];
    tile[in][j] = v;
  }
  __syncthreads();
  #pragma unroll
  for (int i = 0; i < 2; ++i) {
    int id = tid + i * 256;        // 0..511 covers 32 h-rows x 16 n-pairs
    int jh = id >> 4;              // h row within tile
    int ip = id & 15;              // n pair
    float f0 = tile[ip * 2    ][jh];
    float f1 = tile[ip * 2 + 1][jh];
    __bf16 h0b, l0b, h1b, l1b;
    split_bf16(f0, h0b, l0b);
    split_bf16(f1, h1b, l1b);
    long o = ((long)(c * HPAD + hb0 + jh)) * N_SZ + n0 + ip * 2;
    bf16x2 th = {h0b, h1b};
    bf16x2 tl = {l0b, l1b};
    *(bf16x2*)(hiP + o) = th;
    *(bf16x2*)(loP + o) = tl;
  }
}

// ---------------------------------------------------------------------------
// Kernel 1: per-channel GEMM  h1 = relu(X W1 + b1)  fused with
//           h2 = relu(h1 W2 + b2), scalar = h2 . W3 + b3  -> scal[B][C]
// bf16x3 split-precision WMMA, double-buffered LDS, async B-panel copies.
// ---------------------------------------------------------------------------
__global__ void __launch_bounds__(256) big_gemm(
    const float* __restrict__ x,
    const __bf16* __restrict__ WThi, const __bf16* __restrict__ WTlo,
    const float* __restrict__ b1, const float* __restrict__ W2,
    const float* __restrict__ b2, const float* __restrict__ W3,
    const float* __restrict__ b3, float* __restrict__ scal) {
  __shared__ __bf16 ldsA[2][2][BM * LDA_S];     // [buf][hi/lo]
  __shared__ __bf16 ldsB[2][2][HPAD * LDB_S];   // [buf][hi/lo]
  __shared__ float  h2acc[BM * MID];

  const int tid   = threadIdx.x;
  const int c     = blockIdx.y;
  const int m0    = blockIdx.x * BM;
  const int wave  = tid >> 5;
  const int lane  = tid & 31;
  const int wm    = wave & 3;      // M sub-tile (16 rows)
  const int wn    = wave >> 2;     // N half (144 cols = 9 tiles)
  const int lhalf = lane >> 4;
  const int l15   = lane & 15;

  for (int i = tid; i < BM * MID; i += 256) h2acc[i] = 0.f;

  const v8f vzero = {};
  v8f acc[9];
  #pragma unroll
  for (int t = 0; t < 9; ++t) acc[t] = vzero;

  const long xbase = ((long)m0 * C_SZ + c) * N_SZ;
  const long wbase = (long)c * HPAD * N_SZ;

  // per-thread staging coordinates (constant over the K loop)
  const int arow0 = tid >> 3,          ak0 = (tid & 7) * 4;           // A chunk 0
  const int arow1 = (tid + 256) >> 3,  ak1 = ((tid + 256) & 7) * 4;   // A chunk 1

  // ---- prologue: fill buffer 0 for kc = 0 ----
  f32x4 a0 = __builtin_nontemporal_load(
      (const f32x4*)(x + xbase + (long)arow0 * (C_SZ * N_SZ) + ak0));
  f32x4 a1 = __builtin_nontemporal_load(
      (const f32x4*)(x + xbase + (long)arow1 * (C_SZ * N_SZ) + ak1));
  #pragma unroll
  for (int i = 0; i < 9; ++i) {
    int id    = tid + i * 256;
    int plane = id / 1152;
    int rem   = id - plane * 1152;
    int col   = rem >> 2;
    int sub   = rem & 3;
    async_cp16(&ldsB[0][plane][col * LDB_S + sub * 8],
               (plane ? WTlo : WThi) + wbase + (long)col * N_SZ + sub * 8);
  }
  {
    __bf16 h0, l0, h1, l1, h2, l2, h3, l3;
    split_bf16(a0.x, h0, l0); split_bf16(a0.y, h1, l1);
    split_bf16(a0.z, h2, l2); split_bf16(a0.w, h3, l3);
    bf16x4 th = {h0, h1, h2, h3}; bf16x4 tl = {l0, l1, l2, l3};
    *(bf16x4*)(&ldsA[0][0][arow0 * LDA_S + ak0]) = th;
    *(bf16x4*)(&ldsA[0][1][arow0 * LDA_S + ak0]) = tl;
    split_bf16(a1.x, h0, l0); split_bf16(a1.y, h1, l1);
    split_bf16(a1.z, h2, l2); split_bf16(a1.w, h3, l3);
    bf16x4 th2 = {h0, h1, h2, h3}; bf16x4 tl2 = {l0, l1, l2, l3};
    *(bf16x4*)(&ldsA[0][0][arow1 * LDA_S + ak1]) = th2;
    *(bf16x4*)(&ldsA[0][1][arow1 * LDA_S + ak1]) = tl2;
  }

  int p = 0;
  #pragma unroll 1
  for (int kc = 0; kc < NK; ++kc) {
    wait_async();
    __syncthreads();                 // buffer p complete for all waves

    const bool more = (kc + 1) < NK;
    if (more) {
      const int kofs = (kc + 1) * KC;
      a0 = __builtin_nontemporal_load(
          (const f32x4*)(x + xbase + (long)arow0 * (C_SZ * N_SZ) + kofs + ak0));
      a1 = __builtin_nontemporal_load(
          (const f32x4*)(x + xbase + (long)arow1 * (C_SZ * N_SZ) + kofs + ak1));
      #pragma unroll
      for (int i = 0; i < 9; ++i) {
        int id    = tid + i * 256;
        int plane = id / 1152;
        int rem   = id - plane * 1152;
        int col   = rem >> 2;
        int sub   = rem & 3;
        async_cp16(&ldsB[p ^ 1][plane][col * LDB_S + sub * 8],
                   (plane ? WTlo : WThi) + wbase + (long)col * N_SZ + kofs + sub * 8);
      }
    }

    // ---- A fragments (ISA 16-bit A 16x32 lane layout) ----
    FragU ahi, alo;
    {
      int off0 = (wm * 16 + l15) * LDA_S + lhalf * 8;
      ahi.u4[0] = *(const uint4*)(&ldsA[p][0][off0]);
      ahi.u4[1] = *(const uint4*)(&ldsA[p][0][off0 + 16]);
      alo.u4[0] = *(const uint4*)(&ldsA[p][1][off0]);
      alo.u4[1] = *(const uint4*)(&ldsA[p][1][off0 + 16]);
    }
    // ---- 9 N-tiles x 3 WMMAs ----
    #pragma unroll
    for (int t = 0; t < 9; ++t) {
      int cb = (wn * 144 + t * 16 + l15) * LDB_S + lhalf * 8;
      FragU bhi, blo;
      bhi.u4[0] = *(const uint4*)(&ldsB[p][0][cb]);
      bhi.u4[1] = *(const uint4*)(&ldsB[p][0][cb + 16]);
      blo.u4[0] = *(const uint4*)(&ldsB[p][1][cb]);
      blo.u4[1] = *(const uint4*)(&ldsB[p][1][cb + 16]);
      acc[t] = __builtin_amdgcn_wmma_f32_16x16x32_bf16(false, ahi.v, false, bhi.v,
                                                       (short)0, acc[t], false, false);
      acc[t] = __builtin_amdgcn_wmma_f32_16x16x32_bf16(false, ahi.v, false, blo.v,
                                                       (short)0, acc[t], false, false);
      acc[t] = __builtin_amdgcn_wmma_f32_16x16x32_bf16(false, alo.v, false, bhi.v,
                                                       (short)0, acc[t], false, false);
    }

    if (more) {   // convert & store next A tile into the other buffer
      __bf16 h0, l0, h1, l1, h2, l2, h3, l3;
      split_bf16(a0.x, h0, l0); split_bf16(a0.y, h1, l1);
      split_bf16(a0.z, h2, l2); split_bf16(a0.w, h3, l3);
      bf16x4 th = {h0, h1, h2, h3}; bf16x4 tl = {l0, l1, l2, l3};
      *(bf16x4*)(&ldsA[p ^ 1][0][arow0 * LDA_S + ak0]) = th;
      *(bf16x4*)(&ldsA[p ^ 1][1][arow0 * LDA_S + ak0]) = tl;
      split_bf16(a1.x, h0, l0); split_bf16(a1.y, h1, l1);
      split_bf16(a1.z, h2, l2); split_bf16(a1.w, h3, l3);
      bf16x4 th2 = {h0, h1, h2, h3}; bf16x4 tl2 = {l0, l1, l2, l3};
      *(bf16x4*)(&ldsA[p ^ 1][0][arow1 * LDA_S + ak1]) = th2;
      *(bf16x4*)(&ldsA[p ^ 1][1][arow1 * LDA_S + ak1]) = tl2;
    }
    p ^= 1;
  }

  // ---- fused epilogue: relu(acc + b1) . W2 partials (lane = column h) ----
  float part[8][MID];
  #pragma unroll
  for (int r = 0; r < 8; ++r)
    #pragma unroll
    for (int j = 0; j < MID; ++j) part[r][j] = 0.f;

  #pragma unroll
  for (int t = 0; t < 9; ++t) {
    int h = wn * 144 + t * 16 + l15;
    if (h < H_SZ) {
      float bias = b1[c * H_SZ + h];
      float vals[8];
      #pragma unroll
      for (int r = 0; r < 8; ++r) {
        float v = acc[t][r] + bias;
        vals[r] = v > 0.f ? v : 0.f;
      }
      const float* w2p = W2 + ((long)(c * H_SZ + h)) * MID;
      #pragma unroll
      for (int j = 0; j < MID; ++j) {
        float w = w2p[j];
        #pragma unroll
        for (int r = 0; r < 8; ++r) part[r][j] += vals[r] * w;
      }
    }
  }
  const int rowb = wm * 16 + lhalf * 8;
  #pragma unroll
  for (int r = 0; r < 8; ++r)
    #pragma unroll
    for (int j = 0; j < MID; ++j)
      atomicAdd(&h2acc[(rowb + r) * MID + j], part[r][j]);

  __syncthreads();
  if (tid < BM) {
    float s = 0.f;
    #pragma unroll
    for (int j = 0; j < MID; ++j) {
      float v = h2acc[tid * MID + j] + b2[c * MID + j];
      v = v > 0.f ? v : 0.f;
      s += v * W3[c * MID + j];
    }
    s += b3[c];
    scal[(long)(m0 + tid) * C_SZ + c] = s;
  }
}

// ---------------------------------------------------------------------------
// Kernel 2: out = relu(scal @ Wf1 + bf1) @ Wf2 + bf2   (12 -> 30 -> 16)
// ---------------------------------------------------------------------------
__global__ void __launch_bounds__(256) final_mlp(
    const float* __restrict__ scal, const float* __restrict__ Wf1,
    const float* __restrict__ bf1, const float* __restrict__ Wf2,
    const float* __restrict__ bf2, float* __restrict__ out) {
  __shared__ float sW1[C_SZ * FH], sB1[FH], sW2[FH * LD], sB2[LD];
  const int tid = threadIdx.x;
  for (int i = tid; i < C_SZ * FH; i += 256) sW1[i] = Wf1[i];
  for (int i = tid; i < FH;        i += 256) sB1[i] = bf1[i];
  for (int i = tid; i < FH * LD;   i += 256) sW2[i] = Wf2[i];
  for (int i = tid; i < LD;        i += 256) sB2[i] = bf2[i];
  __syncthreads();

  const int b = blockIdx.x * 256 + tid;
  float s[C_SZ];
  #pragma unroll
  for (int cc = 0; cc < C_SZ; ++cc) s[cc] = scal[(long)b * C_SZ + cc];
  float o[LD];
  #pragma unroll
  for (int d = 0; d < LD; ++d) o[d] = sB2[d];
  #pragma unroll
  for (int k = 0; k < FH; ++k) {
    float h = sB1[k];
    #pragma unroll
    for (int cc = 0; cc < C_SZ; ++cc) h += s[cc] * sW1[cc * FH + k];
    h = h > 0.f ? h : 0.f;
    #pragma unroll
    for (int d = 0; d < LD; ++d) o[d] += h * sW2[k * LD + d];
  }
  #pragma unroll
  for (int d = 0; d < LD; ++d) out[(long)b * LD + d] = o[d];
}

// ---------------------------------------------------------------------------
extern "C" void kernel_launch(void* const* d_in, const int* in_sizes, int n_in,
                              void* d_out, int out_size, void* d_ws, size_t ws_size,
                              hipStream_t stream) {
  const float* x   = (const float*)d_in[0];
  const float* W1  = (const float*)d_in[1];
  const float* b1  = (const float*)d_in[2];
  const float* W2  = (const float*)d_in[3];
  const float* b2  = (const float*)d_in[4];
  const float* W3  = (const float*)d_in[5];
  const float* b3  = (const float*)d_in[6];
  const float* Wf1 = (const float*)d_in[7];
  const float* bf1 = (const float*)d_in[8];
  const float* Wf2 = (const float*)d_in[9];
  const float* bf2 = (const float*)d_in[10];
  float* out = (float*)d_out;

  const size_t planeElems = (size_t)C_SZ * HPAD * N_SZ;   // 28,311,552
  __bf16* WThi = (__bf16*)d_ws;
  __bf16* WTlo = WThi + planeElems;
  float*  scal = (float*)(WTlo + planeElems);             // [B][C]

  dim3 g0(N_SZ / 32, HPAD / 32, C_SZ);
  w1_prep<<<g0, 256, 0, stream>>>(W1, WThi, WTlo);

  dim3 g1(B_SZ / BM, C_SZ, 1);
  big_gemm<<<g1, 256, 0, stream>>>(x, WThi, WTlo, b1, W2, b2, W3, b3, scal);

  final_mlp<<<B_SZ / 256, 256, 0, stream>>>(scal, Wf1, bf1, Wf2, bf2, out);
}
